// HybridGNN_1812476199304
// MI455X (gfx1250) — matrix operand order repaired and checked by reading, not compiled, not run on hardware
//
#include <hip/hip_runtime.h>

// ---------------------------------------------------------------------------
// HybridGNN forward for MI455X (gfx1250, wave32, WMMA).
// GEMMs: v_wmma_f32_16x16x32_bf16; weight panel staged in LDS (ds_load_b128,
// bank-conflict-free via row padding), activations streamed from global as
// bf16 b128 loads, all NT B-fragments batched per K-step so loads overlap
// the WMMA pipe. Graph softmax/aggregation are atomic scatter passes (the
// HBM-bound part of this op).
// ---------------------------------------------------------------------------

typedef __attribute__((ext_vector_type(16))) __bf16          v16bf;
typedef __attribute__((ext_vector_type(8)))  float           v8f;
typedef __attribute__((ext_vector_type(8)))  unsigned short  u16x8;

union BF16x16 { v16bf v; u16x8 h[2]; };

static __device__ __forceinline__ unsigned short f2bf(float f) {
    unsigned x = __float_as_uint(f);                       // finite data: plain RNE
    return (unsigned short)((x + 0x7FFFu + ((x >> 16) & 1u)) >> 16);
}

// Monotonic float<->uint encoding for atomicMax on floats (incl. negatives).
static __device__ __forceinline__ unsigned fenc(float f) {
    unsigned u = __float_as_uint(f);
    return (u & 0x80000000u) ? ~u : (u | 0x80000000u);
}
static __device__ __forceinline__ float fdec(unsigned e) {
    return __uint_as_float((e & 0x80000000u) ? (e ^ 0x80000000u) : ~e);
}

// ---------------------------------------------------------------------------
// GEMM: C[M, NT*16] = act(A[M,K] @ W[NT*16, K]^T + bias)
// A, W bf16 row-major (lda/ldw elements, multiples of 16; K%32==0, M%16==0).
// Block = 8 waves; W panel staged into LDS once (pitch K+8 halves: row stride
// 2K+16 B == 16 mod 256 -> 16 rows hit 16 distinct 4-bank groups).
// WMMA 16-bit A layout: lane<16: M=lane, halves = K {0..7,16..23};
//                       lane>=16: M=lane-16, halves = K {8..15,24..31}.
// WMMA 16-bit B layout: N = lane&15, same K order as A.
// C/D layout: VGPR r -> M = 8*(lane>>4)+r, N = lane&15.
// ---------------------------------------------------------------------------
template <int NT>
__global__ __launch_bounds__(256) void gemm_bf16_wmma(
    const unsigned short* __restrict__ A, int lda,
    const unsigned short* __restrict__ W, int ldw,
    float* __restrict__ C, int ldc,
    const float* __restrict__ bias, int M, int K, int act /*0:none,1:relu*/) {
    extern __shared__ unsigned short sW[];                 // [NT*16][K+8]
    const int pitch = K + 8;

    // Cooperative stage of the weight panel into LDS (b128 chunks).
    const int chunksPerRow = K >> 3;
    const int totalChunks  = NT * 16 * chunksPerRow;
    for (int c = threadIdx.x; c < totalChunks; c += 256) {
        const int row = c / chunksPerRow;
        const int kc  = (c - row * chunksPerRow) << 3;
        *(u16x8*)(sW + row * pitch + kc) = *(const u16x8*)(W + (size_t)row * ldw + kc);
    }
    __syncthreads();

    const int lane = threadIdx.x & 31;
    const int wave = threadIdx.x >> 5;
    int m0 = (blockIdx.x * 8 + wave) * 16;
    if (m0 > M - 16) m0 = M - 16;   // tail clamp: duplicate tile, identical values
    const int hi  = lane >> 4;      // 0|1
    const int l16 = lane & 15;

    v8f acc[NT] = {};
    const unsigned short* Arow = A + (size_t)(m0 + l16) * lda;
    const unsigned short* Brow = sW + l16 * pitch + 8 * hi;

    for (int k0 = 0; k0 < K; k0 += 32) {
        if (k0 + 32 < K) __builtin_prefetch(Arow + k0 + 32, 0, 1);  // global_prefetch_b8
        BF16x16 af, bf[NT];
        const int base = k0 + 8 * hi;
        af.h[0] = *(const u16x8*)(Arow + base);            // K base..base+7
        af.h[1] = *(const u16x8*)(Arow + base + 16);       // K base+16..base+23
#pragma unroll
        for (int t = 0; t < NT; ++t) {                     // batch all B fragments
            const unsigned short* p = Brow + t * 16 * pitch + k0;
            bf[t].h[0] = *(const u16x8*)(p);               // ds_load_b128
            bf[t].h[1] = *(const u16x8*)(p + 16);
        }
#pragma unroll
        for (int t = 0; t < NT; ++t)                       // then the WMMA chain
            acc[t] = __builtin_amdgcn_wmma_f32_16x16x32_bf16(
                false, af.v, false, bf[t].v, (short)0, acc[t], false, false);
    }
#pragma unroll
    for (int t = 0; t < NT; ++t) {
        const int col = t * 16 + l16;
        const float bv = bias ? bias[col] : 0.f;
#pragma unroll
        for (int r = 0; r < 8; ++r) {
            const int row = m0 + 8 * hi + r;
            float v = acc[t][r] + bv;
            if (act == 1) v = v > 0.f ? v : 0.f;
            C[(size_t)row * ldc + col] = v;
        }
    }
}

// ---------------------------------------------------------------------------
// Small helper kernels
// ---------------------------------------------------------------------------
__global__ void fill_f(float* p, size_t n, float v) {
    size_t i = (size_t)blockIdx.x * blockDim.x + threadIdx.x;
    if (i < n) p[i] = v;
}
__global__ void fill_u(unsigned* p, size_t n, unsigned v) {
    size_t i = (size_t)blockIdx.x * blockDim.x + threadIdx.x;
    if (i < n) p[i] = v;
}
__global__ void cvt_bf16(const float* __restrict__ src, unsigned short* __restrict__ dst,
                         size_t n) {
    size_t i = (size_t)blockIdx.x * blockDim.x + threadIdx.x;
    if (i < n) dst[i] = f2bf(src[i]);
}

// x[uid] = [user_emb, 0];  x[NU+iid] = [item_emb, ctmp_row]
__global__ void build_x(const int* __restrict__ uid, const int* __restrict__ iid,
                        const float* __restrict__ uw, const float* __restrict__ iw,
                        const float* __restrict__ ctmp, float* __restrict__ x,
                        int B, int NU) {
    int idx = blockIdx.x * blockDim.x + threadIdx.x;
    if (idx >= B * 64) return;
    int b = idx >> 6, c = idx & 63;
    int u = uid[b], it = iid[b];
    x[(size_t)u * 128 + c]               = uw[(size_t)u * 64 + c];
    x[(size_t)u * 128 + 64 + c]          = 0.f;
    x[(size_t)(NU + it) * 128 + c]       = iw[(size_t)it * 64 + c];
    x[(size_t)(NU + it) * 128 + 64 + c]  = ctmp[(size_t)b * 64 + c];
}

// vf[h,k] = sum_c att[h,c] * lin[(h*64+c), k]   (folds the lin_dst GEMM away)
__global__ void fold_dst(const float* __restrict__ lin, const float* __restrict__ att,
                         float* __restrict__ vf, int HK) {
    int idx = blockIdx.x * blockDim.x + threadIdx.x;
    if (idx >= HK) return;
    int h = idx >> 7, k = idx & 127;     // K = 128 for both convs
    float s = 0.f;
    for (int c = 0; c < 64; ++c) s += att[h * 64 + c] * lin[(size_t)(h * 64 + c) * 128 + k];
    vf[idx] = s;
}

// a_s[n,h] = sum_c xs[n, h*64+c] * att[h,c]
__global__ void att_scores(const float* __restrict__ xs, const float* __restrict__ att,
                           float* __restrict__ out, int Nn, int hShift) {
    int idx = blockIdx.x * blockDim.x + threadIdx.x;
    if (idx >= (Nn << hShift)) return;
    int h = idx & ((1 << hShift) - 1);
    int n = idx >> hShift;
    const float* row = xs + ((size_t)n << (6 + hShift)) + h * 64;
    const float* a = att + h * 64;
    float s = 0.f;
    for (int c = 0; c < 64; ++c) s += row[c] * a[c];
    out[idx] = s;
}

// a_d[n,h] = x[n,:] . vf[h,:]   (K=128)
__global__ void matvec_scores(const float* __restrict__ X, const float* __restrict__ vf,
                              float* __restrict__ out, int Nn, int hShift) {
    int idx = blockIdx.x * blockDim.x + threadIdx.x;
    if (idx >= (Nn << hShift)) return;
    int h = idx & ((1 << hShift) - 1);
    int n = idx >> hShift;
    const float* row = X + (size_t)n * 128;
    const float* v = vf + h * 128;
    float s = 0.f;
    for (int k = 0; k < 128; ++k) s += row[k] * v[k];
    out[idx] = s;
}

static __device__ __forceinline__ void edge_sd(const int* ei, int e, int NE, int& s, int& d) {
    if (e < NE) { s = ei[e]; d = ei[NE + e]; } else { s = d = e - NE; }  // self-loop
}

__global__ void edge_alpha_max(const int* __restrict__ ei, int NE, int Etot, int hShift,
                               const float* __restrict__ as_, const float* __restrict__ ad_,
                               float* __restrict__ alpha, unsigned* __restrict__ amax) {
    unsigned idx = blockIdx.x * blockDim.x + threadIdx.x;
    if (idx >= ((unsigned)Etot << hShift)) return;
    int e = idx >> hShift, h = idx & ((1 << hShift) - 1), H = 1 << hShift;
    int s, d; edge_sd(ei, e, NE, s, d);
    float a = as_[s * H + h] + ad_[d * H + h];
    a = a > 0.f ? a : 0.2f * a;                            // leaky_relu(0.2)
    alpha[idx] = a;
    atomicMax(&amax[d * H + h], fenc(a));
}

__global__ void edge_exp_sum(const int* __restrict__ ei, int NE, int Etot, int hShift,
                             float* __restrict__ alpha, const unsigned* __restrict__ amax,
                             float* __restrict__ denom) {
    unsigned idx = blockIdx.x * blockDim.x + threadIdx.x;
    if (idx >= ((unsigned)Etot << hShift)) return;
    int e = idx >> hShift, h = idx & ((1 << hShift) - 1), H = 1 << hShift;
    int s, d; edge_sd(ei, e, NE, s, d);
    float ex = __expf(alpha[idx] - fdec(amax[d * H + h]));
    alpha[idx] = ex;
    atomicAdd(&denom[d * H + h], ex);
}

// out[dst, comp] += xs[src, comp] * softmax_w ; one thread per (edge, channel)
__global__ void edge_scatter(const int* __restrict__ ei, int NE, int Etot, int hcShift,
                             const float* __restrict__ alpha, const float* __restrict__ denom,
                             const float* __restrict__ xs, float* __restrict__ out) {
    size_t idx = (size_t)blockIdx.x * blockDim.x + threadIdx.x;
    if (idx >= ((size_t)Etot << hcShift)) return;
    int e = (int)(idx >> hcShift);
    int comp = (int)(idx & ((1u << hcShift) - 1));
    int h = comp >> 6;                                     // C = 64
    int H = 1 << (hcShift - 6);
    int s, d; edge_sd(ei, e, NE, s, d);
    float w = alpha[(size_t)e * H + h] / fmaxf(denom[d * H + h], 1e-16f);
    atomicAdd(&out[((size_t)d << hcShift) + comp], xs[((size_t)s << hcShift) + comp] * w);
}

// h = act(h + bias); optionally also emit bf16 copy for the next WMMA GEMM
__global__ void bias_act(float* __restrict__ h, const float* __restrict__ bias,
                         size_t total, int cShift, int elu,
                         unsigned short* __restrict__ hb) {
    size_t idx = (size_t)blockIdx.x * blockDim.x + threadIdx.x;
    if (idx >= total) return;
    int c = (int)(idx & ((1u << cShift) - 1));
    float v = h[idx] + bias[c];
    if (elu) v = v > 0.f ? v : (__expf(v) - 1.f);
    h[idx] = v;
    if (hb) hb[idx] = f2bf(v);
}

// comb[b] = [user_emb[uid], h2[NU+iid], h2[uid]]  (bf16 output, GEMM-only use)
__global__ void build_comb(const int* __restrict__ uid, const int* __restrict__ iid,
                           const float* __restrict__ uw, const float* __restrict__ h2,
                           unsigned short* __restrict__ comb, int B, int NU) {
    int idx = blockIdx.x * blockDim.x + threadIdx.x;
    if (idx >= B * 64) return;
    int b = idx >> 6, c = idx & 63;
    int u = uid[b], it = iid[b];
    comb[(size_t)b * 192 + c]        = f2bf(uw[(size_t)u * 64 + c]);
    comb[(size_t)b * 192 + 64 + c]   = f2bf(h2[(size_t)(NU + it) * 64 + c]);
    comb[(size_t)b * 192 + 128 + c]  = f2bf(h2[(size_t)u * 64 + c]);
}

__global__ void final_pred(const float* __restrict__ hid, const float* __restrict__ w2,
                           const float* __restrict__ b2, float* __restrict__ out, int B) {
    int b = blockIdx.x * blockDim.x + threadIdx.x;
    if (b >= B) return;
    float s = b2[0];
    for (int c = 0; c < 64; ++c) s += hid[(size_t)b * 64 + c] * w2[c];
    out[b] = s;
}

// ---------------------------------------------------------------------------
static inline unsigned blks(size_t n) { return (unsigned)((n + 255) / 256); }
static inline unsigned smemW(int NT, int K) { return (unsigned)(NT * 16 * (K + 8) * 2); }

extern "C" void kernel_launch(void* const* d_in, const int* in_sizes, int n_in,
                              void* d_out, int out_size, void* d_ws, size_t ws_size,
                              hipStream_t stream) {
    const int*   user_ids = (const int*)d_in[0];
    const int*   item_ids = (const int*)d_in[1];
    const float* content  = (const float*)d_in[2];
    const int*   eidx     = (const int*)d_in[3];
    const float* uw       = (const float*)d_in[4];
    const float* iw       = (const float*)d_in[5];
    const float* cw       = (const float*)d_in[6];
    const float* cb       = (const float*)d_in[7];
    const float* c1_ls    = (const float*)d_in[8];
    const float* c1_ld    = (const float*)d_in[9];
    const float* c1_as    = (const float*)d_in[10];
    const float* c1_ad    = (const float*)d_in[11];
    const float* c1_b     = (const float*)d_in[12];
    const float* c2_ls    = (const float*)d_in[13];
    const float* c2_ld    = (const float*)d_in[14];
    const float* c2_as    = (const float*)d_in[15];
    const float* c2_ad    = (const float*)d_in[16];
    const float* c2_b     = (const float*)d_in[17];
    const float* pw1      = (const float*)d_in[18];
    const float* pb1      = (const float*)d_in[19];
    const float* pw2      = (const float*)d_in[20];
    const float* pb2      = (const float*)d_in[21];

    const int B    = in_sizes[0];
    const int CFD  = in_sizes[2] / B;
    const int NE   = in_sizes[3] / 2;
    const int NU   = in_sizes[4] / 64;
    const int NI   = in_sizes[5] / 64;
    const int Nn   = NU + NI;
    const int Etot = NE + Nn;

    float* ws = (float*)d_ws;
    size_t o = 0;
    auto alloc = [&](size_t nfloats) {
        size_t r = o; o += (nfloats + 63) & ~(size_t)63; return r;
    };
    const size_t oX    = alloc((size_t)Nn * 128);   // f32 node features
    const size_t oXB   = alloc((size_t)Nn * 64);    // bf16 copy of X (N*128 ushort)
    const size_t oXS1  = alloc((size_t)Nn * 128);   // conv1 xs (f32)
    const size_t oH1   = alloc((size_t)Nn * 128);   // conv1 out (f32)
    const size_t oH1B  = alloc((size_t)Nn * 64);    // bf16 copy of H1
    const size_t oXS2  = alloc((size_t)Nn * 64);    // conv2 xs (f32)
    const size_t oH2   = alloc((size_t)Nn * 64);    // conv2 out (f32)
    const size_t oAL   = alloc((size_t)Etot * 2);   // per-edge alpha/exp
    const size_t oAS1  = alloc((size_t)Nn * 2);
    const size_t oAD1  = alloc((size_t)Nn * 2);
    const size_t oMX1  = alloc((size_t)Nn * 2);
    const size_t oDN1  = alloc((size_t)Nn * 2);
    const size_t oAS2  = alloc(Nn);
    const size_t oAD2  = alloc(Nn);
    const size_t oMX2  = alloc(Nn);
    const size_t oDN2  = alloc(Nn);
    const size_t oVF1  = alloc(2 * 128);
    const size_t oVF2  = alloc(128);
    const size_t oCWB  = alloc((size_t)64 * CFD / 2 + 32);   // bf16 weights
    const size_t oL1B  = alloc(128 * 128 / 2);
    const size_t oL2B  = alloc(64 * 128 / 2);
    const size_t oPW1B = alloc(64 * 192 / 2);

    float* X   = ws + oX;
    float* XS1 = ws + oXS1;
    float* H1  = ws + oH1;
    float* XS2 = ws + oXS2;
    float* H2  = ws + oH2;
    float* AL  = ws + oAL;
    float* AS1 = ws + oAS1;  float* AD1 = ws + oAD1;  float* DN1 = ws + oDN1;
    float* AS2 = ws + oAS2;  float* AD2 = ws + oAD2;  float* DN2 = ws + oDN2;
    float* VF1 = ws + oVF1;  float* VF2 = ws + oVF2;
    unsigned* MX1 = (unsigned*)(ws + oMX1);
    unsigned* MX2 = (unsigned*)(ws + oMX2);
    unsigned short* XB   = (unsigned short*)(ws + oXB);
    unsigned short* H1B  = (unsigned short*)(ws + oH1B);
    unsigned short* CWB  = (unsigned short*)(ws + oCWB);
    unsigned short* L1B  = (unsigned short*)(ws + oL1B);
    unsigned short* L2B  = (unsigned short*)(ws + oL2B);
    unsigned short* PW1B = (unsigned short*)(ws + oPW1B);
    // Lifetime-disjoint aliases inside big regions:
    float*          CTMP = H1;                                  // B*64 f32
    unsigned short* CF16 = (unsigned short*)(ws + oH1 + (size_t)B * 64); // B*CFD bf16
    unsigned short* COMB = (unsigned short*)X;                  // B*192 bf16 (X dead)
    float*          HID  = XS2;                                 // B*64 f32 (XS2 dead)

    const dim3 T(256);

    // --- weight + input conversions (bf16) ---------------------------------
    cvt_bf16<<<blks((size_t)64 * CFD), T, 0, stream>>>(cw, CWB, (size_t)64 * CFD);
    cvt_bf16<<<blks(128 * 128), T, 0, stream>>>(c1_ls, L1B, 128 * 128);
    cvt_bf16<<<blks(64 * 128), T, 0, stream>>>(c2_ls, L2B, 64 * 128);
    cvt_bf16<<<blks(64 * 192), T, 0, stream>>>(pw1, PW1B, 64 * 192);
    cvt_bf16<<<blks((size_t)B * CFD), T, 0, stream>>>(content, CF16, (size_t)B * CFD);

    // --- node features -----------------------------------------------------
    fill_f<<<blks((size_t)Nn * 128), T, 0, stream>>>(X, (size_t)Nn * 128, 0.f);
    gemm_bf16_wmma<4><<<dim3((B + 127) / 128), T, smemW(4, CFD), stream>>>(
        CF16, CFD, CWB, CFD, CTMP, 64, cb, B, CFD, 0);
    build_x<<<blks((size_t)B * 64), T, 0, stream>>>(user_ids, item_ids, uw, iw, CTMP, X, B, NU);
    cvt_bf16<<<blks((size_t)Nn * 128), T, 0, stream>>>(X, XB, (size_t)Nn * 128);

    // --- GAT conv1 (H=2) ---------------------------------------------------
    fold_dst<<<blks(256), T, 0, stream>>>(c1_ld, c1_ad, VF1, 2 * 128);
    gemm_bf16_wmma<8><<<dim3((Nn + 127) / 128), T, smemW(8, 128), stream>>>(
        XB, 128, L1B, 128, XS1, 128, nullptr, Nn, 128, 0);
    att_scores<<<blks((size_t)Nn * 2), T, 0, stream>>>(XS1, c1_as, AS1, Nn, 1);
    matvec_scores<<<blks((size_t)Nn * 2), T, 0, stream>>>(X, VF1, AD1, Nn, 1);
    fill_u<<<blks((size_t)Nn * 2), T, 0, stream>>>(MX1, (size_t)Nn * 2, 0u);
    fill_f<<<blks((size_t)Nn * 2), T, 0, stream>>>(DN1, (size_t)Nn * 2, 0.f);
    fill_f<<<blks((size_t)Nn * 128), T, 0, stream>>>(H1, (size_t)Nn * 128, 0.f);
    edge_alpha_max<<<blks((size_t)Etot * 2), T, 0, stream>>>(eidx, NE, Etot, 1, AS1, AD1, AL, MX1);
    edge_exp_sum<<<blks((size_t)Etot * 2), T, 0, stream>>>(eidx, NE, Etot, 1, AL, MX1, DN1);
    edge_scatter<<<blks((size_t)Etot * 128), T, 0, stream>>>(eidx, NE, Etot, 7, AL, DN1, XS1, H1);
    bias_act<<<blks((size_t)Nn * 128), T, 0, stream>>>(H1, c1_b, (size_t)Nn * 128, 7, 1, H1B);

    // --- GAT conv2 (H=1) ---------------------------------------------------
    fold_dst<<<blks(128), T, 0, stream>>>(c2_ld, c2_ad, VF2, 128);
    gemm_bf16_wmma<4><<<dim3((Nn + 127) / 128), T, smemW(4, 128), stream>>>(
        H1B, 128, L2B, 128, XS2, 64, nullptr, Nn, 128, 0);
    att_scores<<<blks(Nn), T, 0, stream>>>(XS2, c2_as, AS2, Nn, 0);
    matvec_scores<<<blks(Nn), T, 0, stream>>>(H1, VF2, AD2, Nn, 0);
    fill_u<<<blks(Nn), T, 0, stream>>>(MX2, (size_t)Nn, 0u);
    fill_f<<<blks(Nn), T, 0, stream>>>(DN2, (size_t)Nn, 0.f);
    fill_f<<<blks((size_t)Nn * 64), T, 0, stream>>>(H2, (size_t)Nn * 64, 0.f);
    edge_alpha_max<<<blks(Etot), T, 0, stream>>>(eidx, NE, Etot, 0, AS2, AD2, AL, MX2);
    edge_exp_sum<<<blks(Etot), T, 0, stream>>>(eidx, NE, Etot, 0, AL, MX2, DN2);
    edge_scatter<<<blks((size_t)Etot * 64), T, 0, stream>>>(eidx, NE, Etot, 6, AL, DN2, XS2, H2);
    bias_act<<<blks((size_t)Nn * 64), T, 0, stream>>>(H2, c2_b, (size_t)Nn * 64, 6, 0, nullptr);

    // --- prediction head ---------------------------------------------------
    build_comb<<<blks((size_t)B * 64), T, 0, stream>>>(user_ids, item_ids, uw, H2, COMB, B, NU);
    gemm_bf16_wmma<4><<<dim3((B + 127) / 128), T, smemW(4, 192), stream>>>(
        COMB, 192, PW1B, 192, HID, 64, pb1, B, 192, 1);
    final_pred<<<blks(B), T, 0, stream>>>(HID, pw2, pb2, (float*)d_out, B);
}
